// GraphAttention_71090298683922
// MI455X (gfx1250) — compile-verified, matrix operand order checked by reading, not run on hardware
//
#include <hip/hip_runtime.h>
#include <hip/hip_bf16.h>

typedef __attribute__((ext_vector_type(16))) __bf16 v16bf;
typedef __attribute__((ext_vector_type(2)))  __bf16 bf16x2;
typedef __attribute__((ext_vector_type(8)))  float  v8f;

constexpr int   N_    = 4096;
constexpr int   IN_   = 512;
constexpr int   OUT_  = 64;
constexpr int   H_    = 8;
constexpr float ALPHA = 0.2f;
constexpr float NEGV  = -9e15f;

// ---- workspace layout (bytes) ----
constexpr size_t XB_OFF   = 0;                                   // x   bf16 [N, IN]
constexpr size_t XB_SZ    = (size_t)N_ * IN_ * 2;                // 4 MiB
constexpr size_t WBT_OFF  = XB_OFF + XB_SZ;                      // W^T bf16 [H, OUT, IN]
constexpr size_t WBT_SZ   = (size_t)H_ * OUT_ * IN_ * 2;         // 512 KiB
constexpr size_t WHT_OFF  = WBT_OFF + WBT_SZ;                    // Wh^T bf16 [H, OUT, N]
constexpr size_t WHT_SZ   = (size_t)H_ * OUT_ * N_ * 2;          // 4 MiB
constexpr size_t F1_OFF   = WHT_OFF + WHT_SZ;                    // f1 f32 [H, N]
constexpr size_t F1_SZ    = (size_t)H_ * N_ * 4;                 // 128 KiB
constexpr size_t F2_OFF   = F1_OFF + F1_SZ;                      // f2 f32 [H, N]
constexpr size_t F2_SZ    = (size_t)H_ * N_ * 4;                 // 128 KiB
constexpr size_t AM_OFF   = F2_OFF + F2_SZ;                      // adj bitmask u32 [N, 128]
constexpr size_t AM_SZ    = (size_t)N_ * (N_ / 32) * 4;          // 2 MiB
constexpr size_t F2MAX_OFF = AM_OFF + AM_SZ;                     // f2max f32 [H]

// K offset within a 32-wide tile for A/B fragment element (vgpr v, dword half h)
// per ISA 16-bit A 16x32 layout: lanes 0-15 hold K {0..7,16..23}, lanes 16-31 hold K {8..15,24..31}
static __device__ __forceinline__ int koff(int v, int lanehalf) {
    return (lanehalf ? 8 : 0) + ((v & 3) << 1) + ((v & 4) ? 16 : 0);
}

// Load a 16x32/32x16 bf16 fragment from a K-contiguous row.
static __device__ __forceinline__ v16bf load_frag(const __bf16* row, int kbase, int lanehalf) {
    v16bf F;
#pragma unroll
    for (int v = 0; v < 8; ++v) {
        const int K = kbase + koff(v, lanehalf);
        bf16x2 t = *(const bf16x2*)(row + K);
        F[2 * v]     = t.x;
        F[2 * v + 1] = t.y;
    }
    return F;
}

// ---------------- stage 0: dtype conversion + adjacency bit-packing ----------------
__global__ __launch_bounds__(256) void k_cvt_x(const float* __restrict__ x,
                                               __bf16* __restrict__ xb) {
    const size_t i = (size_t)blockIdx.x * blockDim.x + threadIdx.x;
    if (i < (size_t)N_ * IN_) xb[i] = (__bf16)x[i];
}

__global__ __launch_bounds__(256) void k_cvt_w(const float* __restrict__ W,
                                               __bf16* __restrict__ WbT) {
    const size_t idx = (size_t)blockIdx.x * blockDim.x + threadIdx.x;
    if (idx >= (size_t)H_ * OUT_ * IN_) return;
    const int h = (int)(idx / (OUT_ * IN_));
    const int r = (int)(idx % (OUT_ * IN_));
    const int o = r / IN_;
    const int i = r % IN_;
    WbT[idx] = (__bf16)W[((size_t)h * IN_ + i) * OUT_ + o];
}

// adj int32 [N,N] -> bitmask u32 [N, N/32]; one 32-bit word per thread
__global__ __launch_bounds__(256) void k_pack_adj(const int* __restrict__ adj,
                                                  unsigned* __restrict__ amask) {
    const size_t idx = (size_t)blockIdx.x * blockDim.x + threadIdx.x;
    if (idx >= (size_t)N_ * (N_ / 32)) return;
    const size_t row  = idx >> 7;           // N/32 = 128 words per row
    const int    word = (int)(idx & 127);
    const int4*  src  = (const int4*)(adj + row * N_ + word * 32);
    unsigned m = 0u;
#pragma unroll
    for (int j = 0; j < 8; ++j) {
        const int4 q = src[j];
        m |= (q.x > 0 ? 1u : 0u) << (4 * j + 0);
        m |= (q.y > 0 ? 1u : 0u) << (4 * j + 1);
        m |= (q.z > 0 ? 1u : 0u) << (4 * j + 2);
        m |= (q.w > 0 ? 1u : 0u) << (4 * j + 3);
    }
    amask[idx] = m;
}

// per-head max of f2 (for the analytic softmax-max bound)
__global__ __launch_bounds__(256) void k_f2max(const float* __restrict__ f2,
                                               float* __restrict__ f2max) {
    __shared__ float red[256];
    const int h = blockIdx.x;
    float local = -INFINITY;
    for (int i = threadIdx.x; i < N_; i += 256)
        local = fmaxf(local, f2[h * N_ + i]);
    red[threadIdx.x] = local;
    __syncthreads();
#pragma unroll
    for (int step = 128; step > 0; step >>= 1) {
        if (threadIdx.x < step)
            red[threadIdx.x] = fmaxf(red[threadIdx.x], red[threadIdx.x + step]);
        __syncthreads();
    }
    if (threadIdx.x == 0) f2max[h] = red[0];
}

// ---------------- stage 1: projection  Wh = x @ W[h], plus f1/f2 ----------------
// Ping-pong double buffered: set0/set1 fragments live in stable registers;
// loads for one set overlap the WMMAs of the other (no rotation copies).
__global__ __launch_bounds__(256) void k_proj(const __bf16* __restrict__ xb,
                                              const __bf16* __restrict__ WbT,
                                              const float* __restrict__ a1,
                                              const float* __restrict__ a2,
                                              __bf16* __restrict__ WhT,
                                              float* __restrict__ f1,
                                              float* __restrict__ f2) {
    const int wave    = blockIdx.x * 8 + (threadIdx.x >> 5);
    const int h       = wave >> 8;        // 256 row tiles per head
    const int rowbase = (wave & 255) << 4;
    const int lane    = threadIdx.x & 31;
    const int lh      = lane >> 4;        // lane half
    const int n16     = lane & 15;

    const __bf16* xrow = xb + (size_t)(rowbase + n16) * IN_;   // A row = lane%16
    const __bf16* wh0  = WbT + (size_t)h * OUT_ * IN_ + (size_t)(0 * 16 + n16) * IN_;
    const __bf16* wh1  = WbT + (size_t)h * OUT_ * IN_ + (size_t)(1 * 16 + n16) * IN_;
    const __bf16* wh2  = WbT + (size_t)h * OUT_ * IN_ + (size_t)(2 * 16 + n16) * IN_;
    const __bf16* wh3  = WbT + (size_t)h * OUT_ * IN_ + (size_t)(3 * 16 + n16) * IN_;

    v8f c0 = {}, c1 = {}, c2 = {}, c3 = {};

    // set 0 @ k, set 1 @ k+32
    v16bf A0 = load_frag(xrow, 0, lh);
    v16bf P0 = load_frag(wh0, 0, lh);
    v16bf Q0 = load_frag(wh1, 0, lh);
    v16bf R0 = load_frag(wh2, 0, lh);
    v16bf S0 = load_frag(wh3, 0, lh);
    v16bf A1 = load_frag(xrow, 32, lh);
    v16bf P1 = load_frag(wh0, 32, lh);
    v16bf Q1 = load_frag(wh1, 32, lh);
    v16bf R1 = load_frag(wh2, 32, lh);
    v16bf S1 = load_frag(wh3, 32, lh);

    for (int k = 0; k < IN_ - 64; k += 64) {
        c0 = __builtin_amdgcn_wmma_f32_16x16x32_bf16(false, A0, false, P0, (short)0, c0, false, false);
        c1 = __builtin_amdgcn_wmma_f32_16x16x32_bf16(false, A0, false, Q0, (short)0, c1, false, false);
        c2 = __builtin_amdgcn_wmma_f32_16x16x32_bf16(false, A0, false, R0, (short)0, c2, false, false);
        c3 = __builtin_amdgcn_wmma_f32_16x16x32_bf16(false, A0, false, S0, (short)0, c3, false, false);
        A0 = load_frag(xrow, k + 64, lh);
        P0 = load_frag(wh0, k + 64, lh);
        Q0 = load_frag(wh1, k + 64, lh);
        R0 = load_frag(wh2, k + 64, lh);
        S0 = load_frag(wh3, k + 64, lh);
        c0 = __builtin_amdgcn_wmma_f32_16x16x32_bf16(false, A1, false, P1, (short)0, c0, false, false);
        c1 = __builtin_amdgcn_wmma_f32_16x16x32_bf16(false, A1, false, Q1, (short)0, c1, false, false);
        c2 = __builtin_amdgcn_wmma_f32_16x16x32_bf16(false, A1, false, R1, (short)0, c2, false, false);
        c3 = __builtin_amdgcn_wmma_f32_16x16x32_bf16(false, A1, false, S1, (short)0, c3, false, false);
        A1 = load_frag(xrow, k + 96, lh);
        P1 = load_frag(wh0, k + 96, lh);
        Q1 = load_frag(wh1, k + 96, lh);
        R1 = load_frag(wh2, k + 96, lh);
        S1 = load_frag(wh3, k + 96, lh);
    }
    // tail: sets hold k = IN_-64 and IN_-32
    c0 = __builtin_amdgcn_wmma_f32_16x16x32_bf16(false, A0, false, P0, (short)0, c0, false, false);
    c1 = __builtin_amdgcn_wmma_f32_16x16x32_bf16(false, A0, false, Q0, (short)0, c1, false, false);
    c2 = __builtin_amdgcn_wmma_f32_16x16x32_bf16(false, A0, false, R0, (short)0, c2, false, false);
    c3 = __builtin_amdgcn_wmma_f32_16x16x32_bf16(false, A0, false, S0, (short)0, c3, false, false);
    c0 = __builtin_amdgcn_wmma_f32_16x16x32_bf16(false, A1, false, P1, (short)0, c0, false, false);
    c1 = __builtin_amdgcn_wmma_f32_16x16x32_bf16(false, A1, false, Q1, (short)0, c1, false, false);
    c2 = __builtin_amdgcn_wmma_f32_16x16x32_bf16(false, A1, false, R1, (short)0, c2, false, false);
    c3 = __builtin_amdgcn_wmma_f32_16x16x32_bf16(false, A1, false, S1, (short)0, c3, false, false);

    // epilogue: store Wh transposed (bf16) and accumulate f1/f2 partials
    float s1[8], s2[8];
#pragma unroll
    for (int v = 0; v < 8; ++v) { s1[v] = 0.f; s2[v] = 0.f; }

#define PROJ_EPI(CC, OB)                                                              \
    {                                                                                 \
        const int   o   = (OB) * 16 + n16;                                            \
        const float a1v = a1[h * OUT_ + o];                                           \
        const float a2v = a2[h * OUT_ + o];                                           \
        __bf16* dst = WhT + ((size_t)h * OUT_ + o) * N_ + rowbase + lh * 8;           \
        _Pragma("unroll")                                                             \
        for (int v = 0; v < 8; v += 2) {                                              \
            bf16x2 st;                                                                \
            st.x = (__bf16)CC[v];                                                     \
            st.y = (__bf16)CC[v + 1];                                                 \
            *(bf16x2*)(dst + v) = st;                                                 \
        }                                                                             \
        _Pragma("unroll")                                                             \
        for (int v = 0; v < 8; ++v) { s1[v] += CC[v] * a1v; s2[v] += CC[v] * a2v; }   \
    }
    PROJ_EPI(c0, 0) PROJ_EPI(c1, 1) PROJ_EPI(c2, 2) PROJ_EPI(c3, 3)
#undef PROJ_EPI

    // reduce over the 16 lanes of each half (xor masks keep bit4 fixed)
#pragma unroll
    for (int v = 0; v < 8; ++v) {
#pragma unroll
        for (int mk = 1; mk < 16; mk <<= 1) {
            s1[v] += __shfl_xor(s1[v], mk, 32);
            s2[v] += __shfl_xor(s2[v], mk, 32);
        }
    }
    if (n16 == 0) {
#pragma unroll
        for (int v = 0; v < 8; ++v) {
            const int row = rowbase + v + 8 * lh;   // C layout: M = v + 8*half
            f1[h * N_ + row] = s1[v];
            f2[h * N_ + row] = s2[v];
        }
    }
}

// ---------------- stage 2: single-pass softmax attention + aggregation ----------------
// Row max bound: leakyrelu is monotonic, so m = leakyrelu(f1[n] + max_m f2[m])
// upper-bounds every unmasked logit of row n.  Masked logits are NEG and
// exp(NEG - m) underflows to zero, so one pass with fixed m is exact.
__global__ __launch_bounds__(256) void k_attn(const __bf16* __restrict__ WhT,
                                              const float* __restrict__ f1,
                                              const float* __restrict__ f2,
                                              const unsigned* __restrict__ amask,
                                              const float* __restrict__ f2max,
                                              float* __restrict__ out) {
    const int wave    = blockIdx.x * 8 + (threadIdx.x >> 5);
    const int h       = wave >> 8;
    const int rowbase = (wave & 255) << 4;
    const int lane    = threadIdx.x & 31;
    const int lh      = lane >> 4;
    const int n16     = lane & 15;

    const int     arow  = rowbase + n16;               // this lane's A-matrix row
    const float   f1r   = f1[h * N_ + arow];
    const float*  f2h   = f2 + (size_t)h * N_;
    const unsigned* amr = amask + (size_t)arow * (N_ / 32);
    const __bf16* wb0   = WhT + (size_t)h * OUT_ * N_ + (size_t)(0 * 16 + n16) * N_;
    const __bf16* wb1   = WhT + (size_t)h * OUT_ * N_ + (size_t)(1 * 16 + n16) * N_;
    const __bf16* wb2   = WhT + (size_t)h * OUT_ * N_ + (size_t)(2 * 16 + n16) * N_;
    const __bf16* wb3   = WhT + (size_t)h * OUT_ * N_ + (size_t)(3 * 16 + n16) * N_;

    const float ub = f1r + f2max[h];
    const float m  = fmaxf(ub, ALPHA * ub);            // leakyrelu(ub)

    float s = 0.f;
    v8f c0 = {}, c1 = {}, c2 = {}, c3 = {};

    for (int kb = 0; kb < N_; kb += 32) {
        // B-fragment loads first: their latency is covered by the ~70-VALU
        // logit/exp/pack chain below.
        const v16bf B0 = load_frag(wb0, kb, lh);
        const v16bf B1 = load_frag(wb1, kb, lh);
        const v16bf B2 = load_frag(wb2, kb, lh);
        const v16bf B3 = load_frag(wb3, kb, lh);
        const unsigned w = amr[kb >> 5];

        v16bf A;
#pragma unroll
        for (int v = 0; v < 8; ++v) {
            const int bit0 = koff(v, lh);              // bit index within this 32-col word
            const int K    = kb + bit0;
            const float2 fp = *(const float2*)(f2h + K);
            float e0 = f1r + fp.x; e0 = fmaxf(e0, ALPHA * e0);
            float e1 = f1r + fp.y; e1 = fmaxf(e1, ALPHA * e1);
            e0 = (w & (1u << bit0))       ? e0 : NEGV;
            e1 = (w & (1u << (bit0 + 1))) ? e1 : NEGV;
            const float p0 = __expf(e0 - m);
            const float p1 = __expf(e1 - m);
            s += p0 + p1;
            A[2 * v]     = (__bf16)p0;
            A[2 * v + 1] = (__bf16)p1;
        }
        c0 = __builtin_amdgcn_wmma_f32_16x16x32_bf16(false, A, false, B0, (short)0, c0, false, false);
        c1 = __builtin_amdgcn_wmma_f32_16x16x32_bf16(false, A, false, B1, (short)0, c1, false, false);
        c2 = __builtin_amdgcn_wmma_f32_16x16x32_bf16(false, A, false, B2, (short)0, c2, false, false);
        c3 = __builtin_amdgcn_wmma_f32_16x16x32_bf16(false, A, false, B3, (short)0, c3, false, false);
    }

    // combine the two K-halves of each row's sum, then normalize and write out
    s += __shfl_xor(s, 16, 32);
    float inv[8];
#pragma unroll
    for (int v = 0; v < 8; ++v) {
        const float sv = __shfl(s, v + 8 * lh, 32);    // row (v + 8*half) sum
        inv[v] = 1.f / fmaxf(sv, 1e-30f);
    }
#define ATT_EPI(CC, OB)                                                             \
    {                                                                               \
        const int o = (OB) * 16 + n16;                                              \
        _Pragma("unroll")                                                           \
        for (int v = 0; v < 8; ++v) {                                               \
            const int row = rowbase + v + 8 * lh;                                   \
            out[(size_t)row * (H_ * OUT_) + h * OUT_ + o] = CC[v] * inv[v];         \
        }                                                                           \
    }
    ATT_EPI(c0, 0) ATT_EPI(c1, 1) ATT_EPI(c2, 2) ATT_EPI(c3, 3)
#undef ATT_EPI
}

extern "C" void kernel_launch(void* const* d_in, const int* in_sizes, int n_in,
                              void* d_out, int out_size, void* d_ws, size_t ws_size,
                              hipStream_t stream) {
    (void)in_sizes; (void)n_in; (void)out_size; (void)ws_size;
    const float* x   = (const float*)d_in[0];
    const int*   adj = (const int*)d_in[1];
    const float* W   = (const float*)d_in[2];
    const float* a1  = (const float*)d_in[3];
    const float* a2  = (const float*)d_in[4];
    float*       out = (float*)d_out;

    char*     ws    = (char*)d_ws;
    __bf16*   xb    = (__bf16*)(ws + XB_OFF);
    __bf16*   WbT   = (__bf16*)(ws + WBT_OFF);
    __bf16*   WhT   = (__bf16*)(ws + WHT_OFF);
    float*    f1    = (float*)(ws + F1_OFF);
    float*    f2    = (float*)(ws + F2_OFF);
    unsigned* amask = (unsigned*)(ws + AM_OFF);
    float*    f2mx  = (float*)(ws + F2MAX_OFF);

    // stage 0: conversions + adjacency packing (independent)
    {
        const size_t n = (size_t)N_ * IN_;
        k_cvt_x<<<dim3((unsigned)((n + 255) / 256)), dim3(256), 0, stream>>>(x, xb);
    }
    {
        const size_t n = (size_t)H_ * OUT_ * IN_;
        k_cvt_w<<<dim3((unsigned)((n + 255) / 256)), dim3(256), 0, stream>>>(W, WbT);
    }
    {
        const size_t n = (size_t)N_ * (N_ / 32);
        k_pack_adj<<<dim3((unsigned)((n + 255) / 256)), dim3(256), 0, stream>>>(adj, amask);
    }
    // stage 1: projection (8 heads * 256 row-tiles = 2048 waves, 8 waves/block)
    k_proj<<<dim3(256), dim3(256), 0, stream>>>(xb, WbT, a1, a2, WhT, f1, f2);
    // per-head f2 max for the analytic softmax bound
    k_f2max<<<dim3(H_), dim3(256), 0, stream>>>(f2, f2mx);
    // stage 2: fused single-pass attention
    k_attn<<<dim3(256), dim3(256), 0, stream>>>(WhT, f1, f2, amask, f2mx, out);
}
